// TNOBlock_55774445306045
// MI455X (gfx1250) — matrix operand order repaired and checked by reading, not compiled, not run on hardware
//
#include <hip/hip_runtime.h>
#include <math.h>

typedef __bf16 bf16_t;
typedef __attribute__((ext_vector_type(16))) __bf16 v16bf;
typedef __attribute__((ext_vector_type(8)))  __bf16 v8bf;
typedef __attribute__((ext_vector_type(4)))  __bf16 v4bf;
typedef __attribute__((ext_vector_type(8)))  float  v8f;

#define BATCH 4
#define SEQ   4096
#define CH    512
#define HID_  2048
#define MROWS (BATCH*SEQ)      // 16384
#define LDQKV 1536

// ---------------------------------------------------------------------------
// WMMA helpers (CDNA5: v_wmma_f32_16x16x32_bf16, wave32)
// ---------------------------------------------------------------------------
__device__ __forceinline__ v8f wmma_bf16(v16bf a, v16bf b, v8f c) {
    return __builtin_amdgcn_wmma_f32_16x16x32_bf16(
        /*neg_a=*/false, a, /*neg_b=*/false, b,
        /*c_mod=*/(short)0, c, /*reuse_a=*/false, /*reuse_b=*/false);
}

union frag_u { v16bf v; v8bf h[2]; };

// A fragment: 16x32 (MxK), row-major source, lane m = lane&15, half = lane>>4.
// Layout: K = e + 16*half (contiguous, matches B fragment).
// -> two contiguous 16-byte vector loads at k0+16*half and k0+16*half+8.
__device__ __forceinline__ v16bf load_a_frag(const bf16_t* __restrict__ A,
                                             int lda, int row0, int k0, int lane) {
    const int half = (lane >> 4) & 1;
    const int m    = lane & 15;
    const bf16_t* p = A + (size_t)(row0 + m) * lda + k0 + half * 16;
    frag_u f;
    f.h[0] = *(const v8bf*)(p);
    f.h[1] = *(const v8bf*)(p + 8);
    return f.v;
}

// B fragment: 32x16 (KxN) where B[k][n] = W[n][k] (W row-major [Nout,K]).
// lane n = lane&15, half = lane>>4 selects K 0..15 / 16..31 (32B contiguous).
__device__ __forceinline__ v16bf load_b_frag(const bf16_t* __restrict__ W,
                                             int ldw, int col0, int k0, int lane) {
    const int half = (lane >> 4) & 1;
    const int n    = lane & 15;
    const bf16_t* p = W + (size_t)(col0 + n) * ldw + k0 + half * 16;
    frag_u f;
    f.h[0] = *(const v8bf*)(p);
    f.h[1] = *(const v8bf*)(p + 8);
    return f.v;
}

// ---------------------------------------------------------------------------
// f32 -> bf16 weight conversion
// ---------------------------------------------------------------------------
__global__ __launch_bounds__(256) void cvt_bf16_kernel(const float* __restrict__ in,
                                                       bf16_t* __restrict__ out, int n) {
    int i = blockIdx.x * 256 + threadIdx.x;
    if (i < n) out[i] = (bf16_t)in[i];
}

// ---------------------------------------------------------------------------
// LayerNorm (+optional pos encoding) -> bf16.  One wave per 512-wide row.
// float4 loads (coalesced 512B per 32 lanes), 8-byte bf16x4 stores.
// ---------------------------------------------------------------------------
__global__ __launch_bounds__(256) void ln_bf16_kernel(const float* __restrict__ x,
                                                      const float* __restrict__ pos,
                                                      const float* __restrict__ w,
                                                      const float* __restrict__ b,
                                                      bf16_t* __restrict__ out) {
    const int row  = blockIdx.x * 8 + (threadIdx.x >> 5);
    const int lane = threadIdx.x & 31;
    const float4* xr = (const float4*)(x + (size_t)row * CH);
    float4 v4[4];
    float s = 0.f, s2 = 0.f;
#pragma unroll
    for (int i = 0; i < 4; ++i) {
        v4[i] = xr[lane + i * 32];
        const float* pv = (const float*)&v4[i];
#pragma unroll
        for (int j = 0; j < 4; ++j) { s += pv[j]; s2 += pv[j] * pv[j]; }
    }
#pragma unroll
    for (int off = 16; off; off >>= 1) {
        s  += __shfl_xor(s,  off, 32);
        s2 += __shfl_xor(s2, off, 32);
    }
    const float mean = s * (1.f / CH);
    const float var  = s2 * (1.f / CH) - mean * mean;
    const float rstd = rsqrtf(var + 1e-5f);
    bf16_t* orow = out + (size_t)row * CH;
#pragma unroll
    for (int i = 0; i < 4; ++i) {
        const int c4 = lane + i * 32;
        float4 wv = ((const float4*)w)[c4];
        float4 bv = ((const float4*)b)[c4];
        float4 pv4 = pos ? ((const float4*)(pos + (size_t)row * CH))[c4]
                         : make_float4(0.f, 0.f, 0.f, 0.f);
        const float* pv = (const float*)&v4[i];
        const float* pw = (const float*)&wv;
        const float* pb = (const float*)&bv;
        const float* pp = (const float*)&pv4;
        v4bf o;
#pragma unroll
        for (int j = 0; j < 4; ++j)
            o[j] = (bf16_t)((pv[j] - mean) * rstd * pw[j] + pb[j] + pp[j]);
        *(v4bf*)(orow + c4 * 4) = o;
    }
}

// ---------------------------------------------------------------------------
// Generic bf16 WMMA GEMM:  Out[M,Nout] = A[M,K] @ W[Nout,K]^T  (+ epilogue)
//   EPI 0: (+bias) -> bf16 out
//   EPI 1: (+bias, exact GELU) -> bf16 out
//   EPI 2: (+bias, +residual f32) -> f32 out
// Block: 256 thr = 8 waves (2x4); wave tile 32x64 (2x4 accumulators,
// 6 fragment loads -> 8 WMMAs per K-step); block tile 64x256.
// K is compile-time; K-loop double-buffered (loads k+32 overlap 8 WMMAs at k).
// ---------------------------------------------------------------------------
template <int EPI, int K>
__global__ __launch_bounds__(256) void gemm_bf16_kernel(
    const bf16_t* __restrict__ A, int lda,
    const bf16_t* __restrict__ W, int ldw,
    const float* __restrict__ bias,
    const float* __restrict__ resid, int ldr,
    void* __restrict__ out, int ldo) {
    const int tid  = threadIdx.x;
    const int lane = tid & 31;
    const int wave = tid >> 5;
    const int row0 = blockIdx.x * 64  + (wave >> 2) * 32;
    const int col0 = blockIdx.y * 256 + (wave & 3)  * 64;

    v8f acc[2][4] = {};
    v16bf a0 = load_a_frag(A, lda, row0,      0, lane);
    v16bf a1 = load_a_frag(A, lda, row0 + 16, 0, lane);
    v16bf b0 = load_b_frag(W, ldw, col0,      0, lane);
    v16bf b1 = load_b_frag(W, ldw, col0 + 16, 0, lane);
    v16bf b2 = load_b_frag(W, ldw, col0 + 32, 0, lane);
    v16bf b3 = load_b_frag(W, ldw, col0 + 48, 0, lane);

    for (int k0 = 32; k0 < K; k0 += 32) {
        v16bf na0 = load_a_frag(A, lda, row0,      k0, lane);
        v16bf na1 = load_a_frag(A, lda, row0 + 16, k0, lane);
        v16bf nb0 = load_b_frag(W, ldw, col0,      k0, lane);
        v16bf nb1 = load_b_frag(W, ldw, col0 + 16, k0, lane);
        v16bf nb2 = load_b_frag(W, ldw, col0 + 32, k0, lane);
        v16bf nb3 = load_b_frag(W, ldw, col0 + 48, k0, lane);
        acc[0][0] = wmma_bf16(a0, b0, acc[0][0]);
        acc[0][1] = wmma_bf16(a0, b1, acc[0][1]);
        acc[0][2] = wmma_bf16(a0, b2, acc[0][2]);
        acc[0][3] = wmma_bf16(a0, b3, acc[0][3]);
        acc[1][0] = wmma_bf16(a1, b0, acc[1][0]);
        acc[1][1] = wmma_bf16(a1, b1, acc[1][1]);
        acc[1][2] = wmma_bf16(a1, b2, acc[1][2]);
        acc[1][3] = wmma_bf16(a1, b3, acc[1][3]);
        a0 = na0; a1 = na1; b0 = nb0; b1 = nb1; b2 = nb2; b3 = nb3;
    }
    acc[0][0] = wmma_bf16(a0, b0, acc[0][0]);
    acc[0][1] = wmma_bf16(a0, b1, acc[0][1]);
    acc[0][2] = wmma_bf16(a0, b2, acc[0][2]);
    acc[0][3] = wmma_bf16(a0, b3, acc[0][3]);
    acc[1][0] = wmma_bf16(a1, b0, acc[1][0]);
    acc[1][1] = wmma_bf16(a1, b1, acc[1][1]);
    acc[1][2] = wmma_bf16(a1, b2, acc[1][2]);
    acc[1][3] = wmma_bf16(a1, b3, acc[1][3]);

    const int mBase = (lane >> 4) << 3;
#pragma unroll
    for (int ti = 0; ti < 2; ++ti) {
#pragma unroll
        for (int tj = 0; tj < 4; ++tj) {
            const int col = col0 + tj * 16 + (lane & 15);
            const float bv = bias ? bias[col] : 0.f;
#pragma unroll
            for (int r = 0; r < 8; ++r) {
                const int row = row0 + ti * 16 + r + mBase;
                float v = acc[ti][tj][r] + bv;
                if (EPI == 1) {
                    v = 0.5f * v * (1.0f + erff(v * 0.70710678118654752f));
                }
                if (EPI == 2) {
                    float o = v + resid[(size_t)row * ldr + col];
                    ((float*)out)[(size_t)row * ldo + col] = o;
                } else {
                    ((bf16_t*)out)[(size_t)row * ldo + col] = (bf16_t)v;
                }
            }
        }
    }
}

// ---------------------------------------------------------------------------
// Transpose V:  qkvb[:, 1024:1536] (per batch [N, D]) -> vt [B, D, N]
// LDS 64x64 tile, both global sides coalesced; pitch 66 avoids bank conflicts.
// ---------------------------------------------------------------------------
__global__ __launch_bounds__(256) void transpose_v_kernel(
    const bf16_t* __restrict__ qkvb,   // [B*N, 1536]
    bf16_t* __restrict__ vt) {         // [B, CH, SEQ]
    const int bid = blockIdx.x;
    const int b   = bid >> 9;               // (SEQ/64)*(CH/64) = 64*8 = 512
    const int t   = bid & 511;
    const int n0  = (t >> 3) * 64;
    const int d0  = (t & 7)  * 64;
    __shared__ bf16_t tile[64][66];
    const int tx = threadIdx.x & 31;
    const int ty = threadIdx.x >> 5;
#pragma unroll
    for (int i = 0; i < 8; ++i) {
        const int n = ty + i * 8;
        const bf16_t* src = qkvb + (size_t)(b * SEQ + n0 + n) * LDQKV + 1024 + d0;
        tile[n][tx]      = src[tx];
        tile[n][tx + 32] = src[tx + 32];
    }
    __syncthreads();
#pragma unroll
    for (int i = 0; i < 8; ++i) {
        const int d = ty + i * 8;
        bf16_t* dst = vt + ((size_t)b * CH + d0 + d) * SEQ + n0;
        dst[tx]      = tile[tx][d];
        dst[tx + 32] = tile[tx + 32][d];
    }
}

// ---------------------------------------------------------------------------
// Flash attention (H=1, D=512).  One block per (batch, 16-query tile).
// 8 waves each own a 64-wide d-slice; S reduced across waves in LDS;
// online softmax keeps running (m, l); scores never touch global memory.
// All 8 global fragments (4 K + 4 V) are issued at the top of the iteration:
// V-fragment latency hides behind the S WMMAs + LDS reduce + softmax.
// ---------------------------------------------------------------------------
__global__ __launch_bounds__(256) void flash_attn_kernel(
    const bf16_t* __restrict__ qkv,   // [B, N, 1536]  (q | k | .)
    const bf16_t* __restrict__ vt,    // [B, CH, SEQ]
    const float*  __restrict__ rpb,   // [127,127,1]
    bf16_t* __restrict__ out) {       // [B, N, 512]
    const int b    = blockIdx.x >> 8;          // SEQ/16 = 256 q-tiles / batch
    const int q0   = (blockIdx.x & 255) * 16;
    const int tid  = threadIdx.x;
    const int lane = tid & 31;
    const int wave = tid >> 5;
    const int ds   = wave * 64;                // this wave's d-slice

    const bf16_t* Q   = qkv + (size_t)b * SEQ * LDQKV;
    const bf16_t* Km  = Q + 512;
    const bf16_t* Vtb = vt + (size_t)b * CH * SEQ;

    __shared__ float  sred[8][512];            // per-wave partial S (16KB)
    __shared__ float  sS[16][32];
    __shared__ bf16_t sP[16][32];
    __shared__ float  sM[16], sL[16], sScale[16];

    if (tid < 16) { sM[tid] = -1e30f; sL[tid] = 0.f; }
    __syncthreads();

    // Q fragments for this wave's d-slice, resident the whole kernel
    v16bf aq0 = load_a_frag(Q, LDQKV, q0, ds,      lane);
    v16bf aq1 = load_a_frag(Q, LDQKV, q0, ds + 32, lane);

    v8f Oacc[4] = {};
    const float scale = 0.044194173824159216f;  // 1/sqrt(512)

    for (int kb = 0; kb < SEQ; kb += 32) {
        // ---- issue ALL global fragment loads for this iteration ----
        v16bf bk0 = load_b_frag(Km, LDQKV, kb,      ds,      lane);
        v16bf bk1 = load_b_frag(Km, LDQKV, kb,      ds + 32, lane);
        v16bf bk2 = load_b_frag(Km, LDQKV, kb + 16, ds,      lane);
        v16bf bk3 = load_b_frag(Km, LDQKV, kb + 16, ds + 32, lane);
        v16bf bV0 = load_b_frag(Vtb, SEQ, ds,      kb, lane);  // B[k=key][n=d]
        v16bf bV1 = load_b_frag(Vtb, SEQ, ds + 16, kb, lane);
        v16bf bV2 = load_b_frag(Vtb, SEQ, ds + 32, kb, lane);
        v16bf bV3 = load_b_frag(Vtb, SEQ, ds + 48, kb, lane);
        if (kb + 32 < SEQ) {   // warm L2->L0 for next key block (global_prefetch_b8)
            __builtin_prefetch(Km + (size_t)(kb + 32 + (lane & 15)) * LDQKV + ds, 0, 3);
        }

        // ---- partial S over this wave's d-slice: S[16 q, 32 keys] ----
        v8f s0 = {}, s1 = {};
        s0 = wmma_bf16(aq0, bk0, s0);
        s0 = wmma_bf16(aq1, bk1, s0);
        s1 = wmma_bf16(aq0, bk2, s1);
        s1 = wmma_bf16(aq1, bk3, s1);
#pragma unroll
        for (int r = 0; r < 8; ++r) {
            sred[wave][r * 32 + lane]       = s0[r];
            sred[wave][256 + r * 32 + lane] = s1[r];
        }
        __syncthreads();

        // ---- reduce across 8 waves, apply scale + rel-pos bias ----
        for (int f = tid; f < 512; f += 256) {
            float a = 0.f;
#pragma unroll
            for (int w = 0; w < 8; ++w) a += sred[w][f];
            const int l = f & 31, r = (f >> 5) & 7, t = f >> 8;
            const int m = r + ((l >> 4) << 3);
            const int n = t * 16 + (l & 15);
            const int kcol = kb + n;
            const float bias = rpb[(kcol >> 6) * 127 + (kcol & 63)];
            sS[m][n] = a * scale + bias;
        }
        __syncthreads();

        // ---- online softmax update (wave 0: lane = row + 16*half) ----
        if (wave == 0) {
            const int m  = lane & 15;
            const int hh = (lane >> 4) * 16;
            float mo = sM[m], mx = mo;
#pragma unroll
            for (int j = 0; j < 16; ++j) mx = fmaxf(mx, sS[m][hh + j]);
            mx = fmaxf(mx, __shfl_xor(mx, 16, 32));
            const float corr = __expf(mo - mx);
            float lsum = 0.f;
#pragma unroll
            for (int j = 0; j < 16; ++j) {
                float p = __expf(sS[m][hh + j] - mx);
                lsum += p;
                sP[m][hh + j] = (bf16_t)p;
            }
            lsum += __shfl_xor(lsum, 16, 32);
            if (lane < 16) {
                sM[m] = mx;
                sL[m] = sL[m] * corr + lsum;
                sScale[m] = corr;
            }
        }
        __syncthreads();

        // ---- rescale O accumulators, then O += P @ V-slice ----
        const int mBase = (lane >> 4) << 3;
#pragma unroll
        for (int r = 0; r < 8; ++r) {
            const float c = sScale[r + mBase];
#pragma unroll
            for (int t = 0; t < 4; ++t) Oacc[t][r] *= c;
        }
        v16bf aP = load_a_frag((const bf16_t*)&sP[0][0], 32, 0, 0, lane);
        Oacc[0] = wmma_bf16(aP, bV0, Oacc[0]);
        Oacc[1] = wmma_bf16(aP, bV1, Oacc[1]);
        Oacc[2] = wmma_bf16(aP, bV2, Oacc[2]);
        Oacc[3] = wmma_bf16(aP, bV3, Oacc[3]);
        __syncthreads();   // protect sred/sS/sP for next iteration
    }

    // ---- finalize: O /= l, store bf16 ----
    const int nloc  = lane & 15;
    const int mBase = (lane >> 4) << 3;
#pragma unroll
    for (int r = 0; r < 8; ++r) {
        const int m = r + mBase;
        const float inv = 1.f / sL[m];
#pragma unroll
        for (int t = 0; t < 4; ++t) {
            out[((size_t)b * SEQ + q0 + m) * CH + ds + t * 16 + nloc] =
                (bf16_t)(Oacc[t][r] * inv);
        }
    }
}

// ---------------------------------------------------------------------------
// Host-side orchestration
// ---------------------------------------------------------------------------
extern "C" void kernel_launch(void* const* d_in, const int* in_sizes, int n_in,
                              void* d_out, int out_size, void* d_ws, size_t ws_size,
                              hipStream_t stream) {
    const float* x      = (const float*)d_in[0];
    const float* pos    = (const float*)d_in[1];
    const float* ln1_w  = (const float*)d_in[2];
    const float* ln1_b  = (const float*)d_in[3];
    const float* qkv_w  = (const float*)d_in[4];
    const float* proj_w = (const float*)d_in[5];
    const float* proj_b = (const float*)d_in[6];
    const float* ln2_w  = (const float*)d_in[7];
    const float* ln2_b  = (const float*)d_in[8];
    const float* fc1_w  = (const float*)d_in[9];
    const float* fc1_b  = (const float*)d_in[10];
    const float* fc2_w  = (const float*)d_in[11];
    const float* fc2_b  = (const float*)d_in[12];
    const float* rpb    = (const float*)d_in[13];

    // workspace layout (bytes)
    char* ws = (char*)d_ws;
    size_t off = 0;
    bf16_t* h     = (bf16_t*)(ws + off); off += (size_t)MROWS * CH   * 2;   // LN1+pos
    bf16_t* qkvb  = (bf16_t*)(ws + off); off += (size_t)MROWS * 1536 * 2;   // q|k|v
    bf16_t* vt    = (bf16_t*)(ws + off); off += (size_t)BATCH * CH * SEQ * 2; // V^T
    bf16_t* attno = (bf16_t*)(ws + off); off += (size_t)MROWS * CH   * 2;   // attn out
    float*  x1    = (float*) (ws + off); off += (size_t)MROWS * CH   * 4;   // residual 1
    bf16_t* h2    = (bf16_t*)(ws + off); off += (size_t)MROWS * CH   * 2;   // LN2
    bf16_t* mid   = (bf16_t*)(ws + off); off += (size_t)MROWS * HID_ * 2;   // GELU(fc1)
    bf16_t* wqkv  = (bf16_t*)(ws + off); off += (size_t)1536 * CH  * 2;
    bf16_t* wproj = (bf16_t*)(ws + off); off += (size_t)CH   * CH  * 2;
    bf16_t* wfc1  = (bf16_t*)(ws + off); off += (size_t)HID_ * CH  * 2;
    bf16_t* wfc2  = (bf16_t*)(ws + off); off += (size_t)CH   * HID_* 2;

    // 1) weights -> bf16
    cvt_bf16_kernel<<<(1536 * CH + 255) / 256, 256, 0, stream>>>(qkv_w,  wqkv,  1536 * CH);
    cvt_bf16_kernel<<<(CH * CH + 255) / 256,   256, 0, stream>>>(proj_w, wproj, CH * CH);
    cvt_bf16_kernel<<<(HID_ * CH + 255) / 256, 256, 0, stream>>>(fc1_w,  wfc1,  HID_ * CH);
    cvt_bf16_kernel<<<(CH * HID_ + 255) / 256, 256, 0, stream>>>(fc2_w,  wfc2,  CH * HID_);

    // 2) h = bf16( LN1(x) + pos )
    ln_bf16_kernel<<<MROWS / 8, 256, 0, stream>>>(x, pos, ln1_w, ln1_b, h);

    // 3) qkv = h @ qkv_w^T   (no bias)
    gemm_bf16_kernel<0, CH><<<dim3(MROWS / 64, 1536 / 256), 256, 0, stream>>>(
        h, CH, wqkv, CH, nullptr, nullptr, 0, qkvb, 1536);

    // 4) V -> V^T (per batch), then flash attention
    transpose_v_kernel<<<BATCH * 512, 256, 0, stream>>>(qkvb, vt);
    flash_attn_kernel<<<BATCH * (SEQ / 16), 256, 0, stream>>>(qkvb, vt, rpb, attno);

    // 5) x1 = x + attn_out @ proj_w^T + proj_b
    gemm_bf16_kernel<2, CH><<<dim3(MROWS / 64, CH / 256), 256, 0, stream>>>(
        attno, CH, wproj, CH, proj_b, x, CH, x1, CH);

    // 6) h2 = bf16( LN2(x1) )
    ln_bf16_kernel<<<MROWS / 8, 256, 0, stream>>>(x1, nullptr, ln2_w, ln2_b, h2);

    // 7) mid = GELU( h2 @ fc1_w^T + fc1_b )
    gemm_bf16_kernel<1, CH><<<dim3(MROWS / 64, HID_ / 256), 256, 0, stream>>>(
        h2, CH, wfc1, CH, fc1_b, nullptr, 0, mid, HID_);

    // 8) out = x1 + mid @ fc2_w^T + fc2_b
    gemm_bf16_kernel<2, HID_><<<dim3(MROWS / 64, CH / 256), 256, 0, stream>>>(
        mid, HID_, wfc2, HID_, fc2_b, x1, CH, (float*)d_out, CH);
}
